// SelfAttention_74457553044286
// MI455X (gfx1250) — matrix operand order repaired and checked by reading, not compiled, not run on hardware
//
#include <hip/hip_runtime.h>

typedef __attribute__((ext_vector_type(16))) _Float16 v16h;
typedef __attribute__((ext_vector_type(8)))  _Float16 v8h;
typedef __attribute__((ext_vector_type(8)))  float    v8f;
typedef __attribute__((ext_vector_type(4)))  float    v4f;
typedef __attribute__((ext_vector_type(4)))  unsigned int u32x4;
typedef __attribute__((ext_vector_type(8)))  int i32x8;
typedef __attribute__((ext_vector_type(4)))  int i32x4;

#define B_  2
#define T_  2048
#define H_  8
#define D_  128
#define NROW (B_ * T_)            // 4096
#define HD   (H_ * D_)            // 1024

static constexpr float QK_SCALE = 0.29730177875068026f;  // 128^-0.25

#if defined(__has_builtin)
#if __has_builtin(__builtin_amdgcn_tensor_load_to_lds)
#define USE_TDM 1
#endif
#if __has_builtin(__builtin_amdgcn_s_wait_tensorcnt)
#define HAVE_WAIT_TENSOR 1
#endif
#if __has_builtin(__builtin_amdgcn_wave_barrier)
#define HAVE_WAVE_BARRIER 1
#endif
#endif

#ifdef USE_TDM
#warning "CDNA5 probe: tensor_load_to_lds AVAILABLE - TDM staging path compiled"
#else
#warning "CDNA5 probe: tensor_load_to_lds NOT available - manual LDS staging path compiled"
#endif

// ---------- helpers ----------

static __device__ __forceinline__ v8f wmma_f16(v16h a, v16h b, v8f c) {
    return __builtin_amdgcn_wmma_f32_16x16x32_f16(
        false, a, false, b, (short)0, c, false, false);
}

// intra-wave ordering for per-wave-private LDS round trips (ds ops are in-order per wave)
static __device__ __forceinline__ void wavebar() {
#ifdef HAVE_WAVE_BARRIER
    __builtin_amdgcn_wave_barrier();
    asm volatile("" ::: "memory");
#else
    __syncthreads();
#endif
}

// A operand (16x32 f16), row-major source, leading dim ld.
// lanes 0-15: row=lane, K {0..7,16..23}; lanes 16-31: same rows, K {8..15,24..31}.
static __device__ __forceinline__ v16h load_a16(const _Float16* __restrict__ p,
                                                int ld, int row0, int col0) {
    const int lane = threadIdx.x & 31;
    const int row  = row0 + (lane & 15);
    const int off  = (lane < 16) ? 0 : 8;
    const _Float16* r = p + (long)row * ld + col0 + off;
    union { v16h v; v8h h[2]; } u;
    u.h[0] = *(const v8h*)(r);
    u.h[1] = *(const v8h*)(r + 16);
    return u.v;
}

// B operand (32x16 f16) from column-contiguous storage bt[n][k], leading dim ld.
// lane holds column n=lane&15; lanes 0-15 K {0..7,16..23}; lanes 16-31 K {8..15,24..31}.
static __device__ __forceinline__ v16h load_b16(const _Float16* __restrict__ bt,
                                                int ld, int k0, int col0) {
    const int lane = threadIdx.x & 31;
    const int col  = col0 + (lane & 15);
    const int koff = k0 + ((lane < 16) ? 0 : 8);
    const _Float16* r = bt + (long)col * ld + koff;
    union { v16h v; v8h h[2]; } u;
    u.h[0] = *(const v8h*)(r);
    u.h[1] = *(const v8h*)(r + 16);
    return u.v;
}

// reductions within each 16-lane half (C-tile rows live across 16 lanes)
static __device__ __forceinline__ float rowmax16(float v) {
    for (int m = 1; m <= 8; m <<= 1) v = fmaxf(v, __shfl_xor(v, m, 32));
    return v;
}
static __device__ __forceinline__ float rowsum16(float v) {
    for (int m = 1; m <= 8; m <<= 1) v += __shfl_xor(v, m, 32);
    return v;
}

#ifdef USE_TDM
// 2D TDM load: global row-major (2B elems, row stride stride0 elems) -> LDS.
static __device__ __forceinline__ void tdm_load_2d(unsigned int lds_addr,
                                                   const void* gptr,
                                                   unsigned int tensor_d0,
                                                   unsigned int tensor_d1,
                                                   unsigned int stride0,
                                                   unsigned int tile_d0,
                                                   unsigned int tile_d1) {
    unsigned long long ga = (unsigned long long)gptr;
    u32x4 g0;
    g0[0] = 1u;                                            // count=1, user mode
    g0[1] = lds_addr;                                      // lds_addr [63:32]
    g0[2] = (unsigned int)(ga & 0xffffffffu);              // global_addr lo
    g0[3] = (unsigned int)((ga >> 32) & 0x1ffffffu)        // global_addr hi (57b)
          | 0x80000000u;                                   // type=2 ("image")
    i32x8 g1;
    g1[0] = (int)(1u << 16);                               // data_size=1 (2 bytes)
    g1[1] = (int)((tensor_d0 & 0xffffu) << 16);            // tensor_dim0[15:0]
    g1[2] = (int)(((tensor_d0 >> 16) & 0xffffu) | ((tensor_d1 & 0xffffu) << 16));
    g1[3] = (int)(((tensor_d1 >> 16) & 0xffffu) | ((tile_d0 & 0xffffu) << 16));
    g1[4] = (int)(tile_d1 & 0xffffu);                      // tile_dim1; tile_dim2=0
    g1[5] = (int)stride0;                                  // tensor_dim0_stride lo
    g1[6] = 0;
    g1[7] = 0;
    i32x4 z4 = {0, 0, 0, 0};
#if __clang_major__ >= 23
    i32x8 z8 = {0, 0, 0, 0, 0, 0, 0, 0};
    __builtin_amdgcn_tensor_load_to_lds(g0, g1, z4, z4, z8, 0);
#else
    __builtin_amdgcn_tensor_load_to_lds(g0, g1, z4, z4, 0);
#endif
}
#endif

// ---------- conversion / transpose kernels ----------

__global__ void cvt_f32_to_f16(const float* __restrict__ src, _Float16* __restrict__ dst, int n) {
    int i = blockIdx.x * 256 + threadIdx.x;
    if (i < n) dst[i] = (_Float16)src[i];
}

// dst[c*rows + r] = src[r*cols + c]
__global__ void transpose_f32_to_f16(const float* __restrict__ src, _Float16* __restrict__ dst,
                                     int rows, int cols) {
    int i = blockIdx.x * 256 + threadIdx.x;
    if (i < rows * cols) {
        int r = i / cols, c = i % cols;
        dst[c * rows + r] = (_Float16)src[i];
    }
}

// ---------- QKV projection: one wave computes a 16x128 strip ----------
// W=0: Q [B,H,T,D] scaled; W=1: K [B,H,T,D] scaled; W=2: V^T [B,H,D,T]
// Epilogue re-layouts through LDS so all global stores are b128.

template <int W>
__global__ __launch_bounds__(32)
void proj_kernel(const _Float16* __restrict__ xh, const _Float16* __restrict__ Wt,
                 _Float16* __restrict__ dst) {
    __shared__ _Float16 Sl[2048];             // 4 KB strip staging
    const int lane = threadIdx.x & 31;
    const int r0 = blockIdx.x * 16;           // row tile (in [0,4096))
    const int cb = blockIdx.y * 128;          // 128-wide column strip == one head
    const int hh = cb >> 7;

    v16h a[4];
#pragma unroll
    for (int c = 0; c < 4; ++c) a[c] = load_a16(xh, D_, r0, c * 32);

    const int n  = lane & 15;
    const int hf = lane >> 4;

    for (int j = 0; j < 8; ++j) {
        const int c0 = cb + j * 16;
        v16h bw0 = load_b16(Wt, D_, 0,  c0);
        v16h bw1 = load_b16(Wt, D_, 32, c0);
        v16h bw2 = load_b16(Wt, D_, 64, c0);
        v16h bw3 = load_b16(Wt, D_, 96, c0);
        v8f acc = {};
        acc = wmma_f16(a[0], bw0, acc);
        acc = wmma_f16(a[1], bw1, acc);
        acc = wmma_f16(a[2], bw2, acc);
        acc = wmma_f16(a[3], bw3, acc);

        const int d = j * 16 + n;             // column within strip (= head dim)
#pragma unroll
        for (int i = 0; i < 8; ++i) {
            const int m = i + 8 * hf;
            if (W == 2) Sl[d * 16 + m] = (_Float16)acc[i];                 // [d][t]
            else        Sl[m * 128 + d] = (_Float16)(acc[i] * QK_SCALE);   // [t][d]
        }
    }
    wavebar();

    const int bidx = r0 >> 11, t0 = r0 & (T_ - 1);
    if (W == 2) {
        _Float16* vb = dst + (((long)(bidx * H_ + hh) * D_)) * T_ + t0;
#pragma unroll
        for (int k = 0; k < 8; ++k) {
            const int u = lane + 32 * k;      // 256 units of 16B, [128 d][2 halves]
            const int d = u >> 1, h2 = u & 1;
            *(v8h*)(vb + (long)d * T_ + h2 * 8) = *(const v8h*)(Sl + d * 16 + h2 * 8);
        }
    } else {
        _Float16* qb = dst + (((long)(bidx * H_ + hh) * T_) + t0) * D_;
#pragma unroll
        for (int k = 0; k < 8; ++k) {
            const int u = lane + 32 * k;      // 256 units of 16B, [16 rows][16 chunks]
            const int row = u >> 4, ch = u & 15;
            *(v8h*)(qb + (long)row * D_ + ch * 8) = *(const v8h*)(Sl + row * 128 + ch * 8);
        }
    }
}

// ---------- flash attention ----------
// 4 waves (128 thr) per block share one (b,h); wave w handles query tile qi=4g+w.
// K/V blocks staged once per block (TDM double-buffered when available); all waves
// run the group's max trip count - fully-masked blocks are exact no-ops.

__global__ __launch_bounds__(128)
void attn_kernel(const _Float16* __restrict__ Qb, const _Float16* __restrict__ Kb,
                 const _Float16* __restrict__ Vtb, _Float16* __restrict__ Ab) {
    __shared__ _Float16 Klds[2][32][128];     // 16 KB [key][d]
    __shared__ _Float16 Vlds[2][128][32];     // 16 KB [d][key]
    __shared__ _Float16 Pl[4][16][32];        // 4 KB per-wave P staging

    const int lane = threadIdx.x & 31;
    const int wid  = threadIdx.x >> 5;
    const int g  = blockIdx.x & 31;
    const int h  = (blockIdx.x >> 5) & 7;
    const int b  = blockIdx.x >> 8;
    const int qi = g * 4 + wid;               // this wave's 16-row query tile

    const _Float16* Qh = Qb  + (long)(b * H_ + h) * T_ * D_;
    const _Float16* Kh = Kb  + (long)(b * H_ + h) * T_ * D_;
    const _Float16* Vh = Vtb + (long)(b * H_ + h) * D_ * T_;

    v16h aq[4];
#pragma unroll
    for (int c = 0; c < 4; ++c) aq[c] = load_a16(Qh, D_, qi * 16, c * 32);

    v8f acc[8] = {};
    float mrow[8], lrow[8];
#pragma unroll
    for (int i = 0; i < 8; ++i) { mrow[i] = -3.0e38f; lrow[i] = 0.0f; }

    const int n  = lane & 15;
    const int hf = lane >> 4;
    const int nkb = 2 * g + 2;                // uniform per block (max causal range)

    auto stage = [&](int kb, int buf) {
#ifdef USE_TDM
        asm volatile("s_wait_dscnt 0x0" ::: "memory");
        tdm_load_2d((unsigned int)(size_t)&Klds[buf][0][0],
                    Kh + (long)kb * 32 * D_, D_, T_, D_, /*tile*/ D_, 32);
        tdm_load_2d((unsigned int)(size_t)&Vlds[buf][0][0],
                    Vh + (long)kb * 32, T_, D_, T_, /*tile*/ 32, D_);
#else
        const int tid = threadIdx.x;
        const v8h* ksrc = (const v8h*)(Kh + (long)kb * 32 * D_);
        v8h* kdst = (v8h*)&Klds[buf][0][0];
        for (int u = tid; u < 512; u += 128) kdst[u] = ksrc[u];
        v8h* vdst = (v8h*)&Vlds[buf][0][0];
        for (int u = tid; u < 512; u += 128) {
            int rr = u >> 2, ch = u & 3;
            vdst[u] = *(const v8h*)(Vh + (long)rr * T_ + kb * 32 + ch * 8);
        }
#endif
    };

#ifdef USE_TDM
    if (wid == 0) stage(0, 0);
#else
    stage(0, 0);
#endif

    for (int kb = 0; kb < nkb; ++kb) {
        const int buf = kb & 1;
        const int j0 = kb * 32;

#ifdef USE_TDM
        if (wid == 0) {
            if (kb + 1 < nkb) {
                stage(kb + 1, buf ^ 1);
#ifdef HAVE_WAIT_TENSOR
                __builtin_amdgcn_s_wait_tensorcnt(2);
#else
                asm volatile("s_wait_tensorcnt 0x2" ::: "memory");
#endif
            } else {
#ifdef HAVE_WAIT_TENSOR
                __builtin_amdgcn_s_wait_tensorcnt(0);
#else
                asm volatile("s_wait_tensorcnt 0x0" ::: "memory");
#endif
            }
        }
#else
        if (kb + 1 < nkb) stage(kb + 1, buf ^ 1);
#endif
        __syncthreads();                      // staged data visible to all waves

        const _Float16* Kt = &Klds[buf][0][0];
        const _Float16* Vt = &Vlds[buf][0][0];

        // ---- scores: two 16x16 tiles over K-dim 128 (8 WMMAs) ----
        v8f s0 = {}, s1 = {};
#pragma unroll
        for (int c = 0; c < 4; ++c) {
            v16h b0 = load_b16(Kt, D_, c * 32, 0);
            v16h b1 = load_b16(Kt, D_, c * 32, 16);
            s0 = wmma_f16(aq[c], b0, s0);
            s1 = wmma_f16(aq[c], b1, s1);
        }

        // ---- causal mask (also turns beyond-range blocks into exact no-ops) ----
        if (j0 + 31 > qi * 16) {
#pragma unroll
            for (int i = 0; i < 8; ++i) {
                const int m = qi * 16 + i + 8 * hf;
                if (j0 + n > m)      s0[i] = -3.0e38f;
                if (j0 + 16 + n > m) s1[i] = -3.0e38f;
            }
        }

        // ---- online softmax ----
        float corr[8];
#pragma unroll
        for (int i = 0; i < 8; ++i) {
            const float rm = rowmax16(fmaxf(s0[i], s1[i]));
            const float mn = fmaxf(mrow[i], rm);
            corr[i] = __expf(mrow[i] - mn);
            s0[i] = __expf(s0[i] - mn);
            s1[i] = __expf(s1[i] - mn);
            const float rs = rowsum16(s0[i] + s1[i]);
            lrow[i] = lrow[i] * corr[i] + rs;
            mrow[i] = mn;
        }
#pragma unroll
        for (int t = 0; t < 8; ++t)
#pragma unroll
            for (int i = 0; i < 8; ++i) acc[t][i] *= corr[i];

        // ---- P (C layout) -> per-wave LDS -> A operand ----
#pragma unroll
        for (int i = 0; i < 8; ++i) {
            Pl[wid][i + 8 * hf][n]      = (_Float16)s0[i];
            Pl[wid][i + 8 * hf][16 + n] = (_Float16)s1[i];
        }
        wavebar();
        v16h ap = load_a16(&Pl[wid][0][0], 32, 0, 0);

        // ---- P @ V : 8 WMMAs ----
#pragma unroll
        for (int t = 0; t < 8; ++t) {
            v16h bv = load_b16(Vt, 32, 0, t * 16);
            acc[t] = wmma_f16(ap, bv, acc[t]);
        }
        __syncthreads();                      // everyone done with buf before restage
    }

    // ---- epilogue: normalize, re-layout via (dead) K staging LDS, b128 stores ----
    float inv[8];
#pragma unroll
    for (int i = 0; i < 8; ++i) inv[i] = 1.0f / lrow[i];

    __syncthreads();                          // K/V LDS now reusable
    _Float16* Ol = ((_Float16*)Klds) + wid * 2048;   // 4 KB per wave
#pragma unroll
    for (int t = 0; t < 8; ++t)
#pragma unroll
        for (int i = 0; i < 8; ++i)
            Ol[(i + 8 * hf) * 128 + t * 16 + n] = (_Float16)(acc[t][i] * inv[i]);
    wavebar();

    _Float16* ab = Ab + ((long)(b * T_ + qi * 16)) * HD + h * D_;
#pragma unroll
    for (int k = 0; k < 8; ++k) {
        const int u = lane + 32 * k;          // 256 units: [16 rows][16 chunks]
        const int row = u >> 4, ch = u & 15;
        *(v8h*)(ab + (long)row * HD + ch * 8) = *(const v8h*)(Ol + row * 128 + ch * 8);
    }
}

// ---------- output projection: attn[4096x1024] @ Wo + bo -> out[4096x128] (f32) ----------

__global__ __launch_bounds__(32)
void outproj_kernel(const _Float16* __restrict__ Ab, const _Float16* __restrict__ Woth,
                    const float* __restrict__ bo, float* __restrict__ out) {
    __shared__ float Sf[16][16];
    const int lane = threadIdx.x & 31;
    const int r0 = blockIdx.x * 16;
    const int c0 = blockIdx.y * 16;
    v8f acc = {};
    for (int c = 0; c < 32; c += 2) {
        v16h a0 = load_a16(Ab,   HD, r0,           c * 32);
        v16h b0 = load_b16(Woth, HD, c * 32,       c0);
        v16h a1 = load_a16(Ab,   HD, r0,           (c + 1) * 32);
        v16h b1 = load_b16(Woth, HD, (c + 1) * 32, c0);
        acc = wmma_f16(a0, b0, acc);
        acc = wmma_f16(a1, b1, acc);
    }
    const int n  = lane & 15;
    const int hf = lane >> 4;
    const float bias = bo[c0 + n];
#pragma unroll
    for (int i = 0; i < 8; ++i) Sf[i + 8 * hf][n] = acc[i] + bias;
    wavebar();
#pragma unroll
    for (int k = 0; k < 2; ++k) {
        const int u = lane + 32 * k;          // 64 units of 16B: [16 rows][4 chunks]
        const int row = u >> 2, ch = u & 3;
        *(v4f*)(out + (long)(r0 + row) * D_ + c0 + ch * 4) = *(const v4f*)(&Sf[row][ch * 4]);
    }
}

// ---------- host side ----------

extern "C" void kernel_launch(void* const* d_in, const int* in_sizes, int n_in,
                              void* d_out, int out_size, void* d_ws, size_t ws_size,
                              hipStream_t stream) {
    (void)in_sizes; (void)n_in; (void)out_size; (void)ws_size;

    const float* x  = (const float*)d_in[0];   // [2,2048,128]
    const float* Wq = (const float*)d_in[1];   // [128,1024]
    const float* Wk = (const float*)d_in[2];
    const float* Wv = (const float*)d_in[3];
    const float* Wo = (const float*)d_in[4];   // [1024,128]
    const float* bo = (const float*)d_in[5];   // [128]
    float* out = (float*)d_out;                // [2,2048,128]

    char* p = (char*)d_ws;
    _Float16* xh   = (_Float16*)p; p += (size_t)NROW * D_ * 2;        // 1 MB
    _Float16* Wth  = (_Float16*)p; p += (size_t)3 * HD * D_ * 2;      // 768 KB
    _Float16* Woth = (_Float16*)p; p += (size_t)D_ * HD * 2;          // 256 KB
    _Float16* Qb   = (_Float16*)p; p += (size_t)NROW * HD * 2;        // 8 MB [B,H,T,D]
    _Float16* Kb   = (_Float16*)p; p += (size_t)NROW * HD * 2;        // 8 MB [B,H,T,D]
    _Float16* Vtb  = (_Float16*)p; p += (size_t)NROW * HD * 2;        // 8 MB [B,H,D,T]
    _Float16* Ab   = (_Float16*)p; p += (size_t)NROW * HD * 2;        // 8 MB [4096,1024]

    cvt_f32_to_f16<<<(NROW * D_ + 255) / 256, 256, 0, stream>>>(x, xh, NROW * D_);
    transpose_f32_to_f16<<<(D_ * HD + 255) / 256, 256, 0, stream>>>(Wq, Wth,             D_, HD);
    transpose_f32_to_f16<<<(D_ * HD + 255) / 256, 256, 0, stream>>>(Wk, Wth + HD * D_,   D_, HD);
    transpose_f32_to_f16<<<(D_ * HD + 255) / 256, 256, 0, stream>>>(Wv, Wth + 2*HD*D_,   D_, HD);
    transpose_f32_to_f16<<<(HD * D_ + 255) / 256, 256, 0, stream>>>(Wo, Woth,            HD, D_);

    dim3 pg(NROW / 16, HD / 128);
    proj_kernel<0><<<pg, 32, 0, stream>>>(xh, Wth,             Qb);
    proj_kernel<1><<<pg, 32, 0, stream>>>(xh, Wth + HD * D_,   Kb);
    proj_kernel<2><<<pg, 32, 0, stream>>>(xh, Wth + 2*HD*D_,   Vtb);

    // 512 blocks x 128 threads: each block = one (b,h) and 4 query tiles
    attn_kernel<<<B_ * H_ * (T_ / 64), 128, 0, stream>>>(Qb, Kb, Vtb, Ab);

    outproj_kernel<<<dim3(NROW / 16, D_ / 16), 32, 0, stream>>>(Ab, Woth, bo, out);
}
